// React_Normal_Block_1881195675755
// MI455X (gfx1250) — 8-bit matrix operand order repaired and checked by reading, compiled, not run on hardware
//
#include <hip/hip_runtime.h>
#include <stdint.h>

typedef int   v16i __attribute__((ext_vector_type(16)));
typedef float v8f  __attribute__((ext_vector_type(8)));

static constexpr int BATCH = 32, CH = 256, Hdim = 56, Wdim = 56;
static constexpr int HW    = Hdim * Wdim;
static constexpr int NPIX  = BATCH * HW;
static constexpr int PGROUPS = NPIX / 256;
static constexpr int NKT1 = 36;
static constexpr int NKT2 = 4;
static constexpr float EPSV = 1e-5f;

static constexpr uint8_t FP8_P1 = 0x38;
static constexpr uint8_t FP8_M1 = 0xB8;
__device__ __forceinline__ uint32_t fp8sign(float d) {
  return (d > 0.f) ? (uint32_t)FP8_P1 : ((d < 0.f) ? (uint32_t)FP8_M1 : 0u);
}

static constexpr size_t SZ_ABIN   = (size_t)NPIX * CH;
static constexpr size_t SZ_F32    = (size_t)NPIX * CH * 4;
static constexpr size_t OFF_ABIN1 = 0;
static constexpr size_t OFF_ABIN2 = OFF_ABIN1 + SZ_ABIN;
static constexpr size_t OFF_OUTR  = OFF_ABIN2 + SZ_ABIN;
static constexpr size_t OFF_XNHWC = OFF_OUTR + SZ_F32;
static constexpr size_t OFF_OUTN  = OFF_XNHWC;
static constexpr size_t OFF_WF1   = OFF_XNHWC + SZ_F32;
static constexpr size_t OFF_WF2   = OFF_WF1 + (size_t)16 * NKT1 * 1024;
static constexpr size_t OFF_AL1   = OFF_WF2 + (size_t)16 * NKT2 * 1024;
static constexpr size_t OFF_AL2   = OFF_AL1 + 1024;
static constexpr size_t OFF_ZPG   = OFF_AL2 + 1024;

__global__ void k_zero(uint8_t* zp) {
  *(int4*)(zp + threadIdx.x * 16) = int4{0, 0, 0, 0};
}

__global__ __launch_bounds__(256) void k_binarize(
    const float* __restrict__ x, const float* __restrict__ beta,
    uint8_t* __restrict__ abin, float* __restrict__ xnhwc) {
  __shared__ float tf[32 * 257];
  const int p0  = blockIdx.x * 32;
  const int n   = p0 / HW;
  const int rr0 = p0 % HW;
  {
    int pj = threadIdx.x & 31;
    int c0 = threadIdx.x >> 5;
    for (int c = c0; c < CH; c += 8)
      tf[pj * 257 + c] = x[((size_t)n * CH + c) * HW + rr0 + pj];
  }
  __syncthreads();
  {
    int pj = threadIdx.x >> 3;
    int c0 = (threadIdx.x & 7) * 32;
    size_t base = (size_t)(p0 + pj) * CH + c0;
    for (int i = 0; i < 32; i += 4) {
      uint32_t packed = 0;
      for (int j = 0; j < 4; ++j) {
        int c = c0 + i + j;
        float v = tf[pj * 257 + c];
        xnhwc[base + i + j] = v;
        packed |= fp8sign(v - beta[c]) << (8 * j);
      }
      *(uint32_t*)(abin + base + i) = packed;
    }
  }
}

__global__ void k_alpha(const float* __restrict__ w, float* __restrict__ alpha, int K) {
  int co = blockIdx.x * blockDim.x + threadIdx.x;
  if (co >= CH) return;
  float s = 0.f;
  for (int k = 0; k < K; ++k) s += fabsf(w[(size_t)co * K + k]);
  alpha[co] = s / (float)K;
}

__global__ void k_packw(const float* __restrict__ w, uint8_t* __restrict__ wfrag,
                        int nkt, int conv3x3) {
  int t = blockIdx.x * blockDim.x + threadIdx.x;
  if (t >= 16 * nkt * 32 * 8) return;
  int v    = t & 7;
  int lane = (t >> 3) & 31;
  int kt   = (t >> 8) % nkt;
  int g    = (t >> 8) / nkt;
  int co    = g * 16 + (lane & 15);
  int kbase = (lane >= 16) ? 8 : 0;
  int k0    = kt * 64 + 16 * (v >> 1) + kbase + 4 * (v & 1);
  uint32_t packed = 0;
  for (int b = 0; b < 4; ++b) {
    int k = k0 + b;
    float wv;
    if (conv3x3) {
      int seg = k >> 8, ci = k & 255;
      int kh = seg / 3, kw = seg % 3;
      wv = w[(((size_t)co * CH + ci) * 3 + kh) * 3 + kw];
    } else {
      wv = w[(size_t)co * CH + k];
    }
    packed |= fp8sign(wv) << (8 * b);
  }
  *(uint32_t*)(wfrag + (size_t)t * 4) = packed;
}

__device__ __forceinline__ v16i load_a128(const uint8_t* ap) {
  v16i a;
#pragma unroll
  for (int half = 0; half < 2; ++half) {
#pragma unroll
    for (int u = 0; u < 4; ++u) {
      int2 d = *(const int2*)(ap + half * 64 + (u << 4));
      a[half * 8 + 2 * u]     = d.x;
      a[half * 8 + 2 * u + 1] = d.y;
    }
  }
  return a;
}

__device__ __forceinline__ v16i load_b128lds(const uint8_t* ldsL, int kt64) {
  v16i b;
#pragma unroll
  for (int half = 0; half < 2; ++half) {
    const int4* bp = (const int4*)(ldsL + (size_t)(kt64 + half * 4) * 1024);
    int4 lo = bp[0], hi = bp[1];
    b[half * 8 + 0] = lo.x; b[half * 8 + 1] = lo.y;
    b[half * 8 + 2] = lo.z; b[half * 8 + 3] = lo.w;
    b[half * 8 + 4] = hi.x; b[half * 8 + 5] = hi.y;
    b[half * 8 + 6] = hi.z; b[half * 8 + 7] = hi.w;
  }
  return b;
}

__global__ __launch_bounds__(256) void k_gemm3x3(
    const uint8_t* __restrict__ abin, const uint8_t* __restrict__ wfrag,
    const float* __restrict__ alpha, const float* __restrict__ xnhwc,
    const float* __restrict__ bng, const float* __restrict__ bnb,
    const float* __restrict__ bnm, const float* __restrict__ bnv,
    const float* __restrict__ prg, const float* __restrict__ prs,
    const float* __restrict__ prz, const float* __restrict__ beta2,
    const uint8_t* __restrict__ zpage,
    float* __restrict__ outr, uint8_t* __restrict__ abin2) {
  extern __shared__ uint8_t smem[];
  const int lane  = threadIdx.x & 31;
  const int wave  = threadIdx.x >> 5;
  const int cc    = blockIdx.y;
  const int ptile = blockIdx.x * 8 + wave;
  const int m     = lane & 15;
  const int khalf = (lane >= 16) ? 8 : 0;

  int pA[2], nimgA[2], hhA[2], wwA[2];
#pragma unroll
  for (int t = 0; t < 2; ++t) {
    pA[t]    = ptile * 32 + t * 16 + m;
    nimgA[t] = pA[t] / HW;
    int rem  = pA[t] % HW;
    hhA[t] = rem / Wdim; wwA[t] = rem % Wdim;
  }

  {
    int nt = threadIdx.x >> 6;
    int f  = (threadIdx.x & 63) * 16;
    const uint8_t* src = wfrag + ((size_t)(cc * 4 + nt) * NKT1) * 1024 + f;
    uint8_t* dst = smem + nt * 1024 + f;
    for (int kt = 0; kt < NKT1; ++kt) {
      *(int4*)dst = *(const int4*)src;
      src += 1024; dst += 4096;
    }
  }
  __syncthreads();

  const uint8_t* ldsL = smem + lane * 32;
  v8f acc[2][4] = {};
#pragma unroll
  for (int seg = 0; seg < 9; ++seg) {
    const int kh = seg / 3, kw = seg - kh * 3;
    const uint8_t* ap[2];
#pragma unroll
    for (int t = 0; t < 2; ++t) {
      int yy = hhA[t] + kh - 1, xx = wwA[t] + kw - 1;
      bool valid = ((unsigned)yy < (unsigned)Hdim) & ((unsigned)xx < (unsigned)Wdim);
      const uint8_t* real =
          abin + (((size_t)nimgA[t] * HW + yy * Wdim + xx) << 8) + khalf;
      ap[t] = valid ? real : zpage;
    }
#pragma unroll
    for (int q = 0; q < 2; ++q) {
      v16i a0 = load_a128(ap[0] + q * 128);
      v16i a1 = load_a128(ap[1] + q * 128);
      const int kt64 = (seg * 4 + q * 2) * 4;
#pragma unroll
      for (int nt = 0; nt < 4; ++nt) {
        v16i b = load_b128lds(ldsL, kt64 + nt);
        acc[0][nt] = __builtin_amdgcn_wmma_f32_16x16x128_fp8_fp8(
            a0, b, (short)0, acc[0][nt], false, false);
        acc[1][nt] = __builtin_amdgcn_wmma_f32_16x16x128_fp8_fp8(
            a1, b, (short)0, acc[1][nt], false, false);
      }
    }
  }

#pragma unroll
  for (int nt = 0; nt < 4; ++nt) {
    int co = cc * 64 + nt * 16 + m;
    float al  = alpha[co];
    float inv = bng[co] * rsqrtf(bnv[co] + EPSV);
    float add = bnb[co] - bnm[co] * inv;
    float gam = prg[co], sl = prs[co], ze = prz[co], b2 = beta2[co];
#pragma unroll
    for (int t = 0; t < 2; ++t) {
#pragma unroll
      for (int r = 0; r < 8; ++r) {
        size_t pp = (size_t)(ptile * 32 + t * 16 + r + khalf);
        float yv  = acc[t][nt][r] * al * inv + add;
        float res = yv + xnhwc[pp * CH + co];
        float tt  = res - gam;
        float o   = (tt > 0.f ? tt : sl * tt) + ze;
        outr[pp * CH + co] = o;
        abin2[pp * CH + co] = (uint8_t)fp8sign(o - b2);
      }
    }
  }
}

__global__ __launch_bounds__(256) void k_gemm1x1(
    const uint8_t* __restrict__ abin2, const uint8_t* __restrict__ wfrag,
    const float* __restrict__ alpha, const float* __restrict__ outr,
    const float* __restrict__ bng, const float* __restrict__ bnb,
    const float* __restrict__ bnm, const float* __restrict__ bnv,
    const float* __restrict__ prg, const float* __restrict__ prs,
    const float* __restrict__ prz, float* __restrict__ outn) {
  __shared__ uint8_t smem[16384];
  const int lane  = threadIdx.x & 31;
  const int wave  = threadIdx.x >> 5;
  const int cc    = blockIdx.y;
  const int ptile = blockIdx.x * 8 + wave;
  const int m     = lane & 15;
  const int khalf = (lane >= 16) ? 8 : 0;

  {
    int nt = threadIdx.x >> 6;
    int f  = (threadIdx.x & 63) * 16;
    const uint8_t* src = wfrag + ((size_t)(cc * 4 + nt) * NKT2) * 1024 + f;
    uint8_t* dst = smem + nt * 1024 + f;
    for (int kt = 0; kt < NKT2; ++kt) {
      *(int4*)dst = *(const int4*)src;
      src += 1024; dst += 4096;
    }
  }
  __syncthreads();

  const uint8_t* ldsL = smem + lane * 32;
  const uint8_t* ap0 = abin2 + ((size_t)(ptile * 32 + m) << 8) + khalf;
  const uint8_t* ap1 = ap0 + (16 << 8);
  v8f acc[2][4] = {};
#pragma unroll
  for (int q = 0; q < 2; ++q) {
    v16i a0 = load_a128(ap0 + q * 128);
    v16i a1 = load_a128(ap1 + q * 128);
    const int kt64 = (q * 2) * 4;
#pragma unroll
    for (int nt = 0; nt < 4; ++nt) {
      v16i b = load_b128lds(ldsL, kt64 + nt);
      acc[0][nt] = __builtin_amdgcn_wmma_f32_16x16x128_fp8_fp8(
          a0, b, (short)0, acc[0][nt], false, false);
      acc[1][nt] = __builtin_amdgcn_wmma_f32_16x16x128_fp8_fp8(
          a1, b, (short)0, acc[1][nt], false, false);
    }
  }

#pragma unroll
  for (int nt = 0; nt < 4; ++nt) {
    int co = cc * 64 + nt * 16 + m;
    float al  = alpha[co];
    float inv = bng[co] * rsqrtf(bnv[co] + EPSV);
    float add = bnb[co] - bnm[co] * inv;
    float gam = prg[co], sl = prs[co], ze = prz[co];
#pragma unroll
    for (int t = 0; t < 2; ++t) {
#pragma unroll
      for (int r = 0; r < 8; ++r) {
        size_t pp = (size_t)(ptile * 32 + t * 16 + r + khalf);
        float yv  = acc[t][nt][r] * al * inv + add;
        float res = yv + outr[pp * CH + co];
        float tt  = res - gam;
        float o   = (tt > 0.f ? tt : sl * tt) + ze;
        outn[pp * CH + co] = o;
      }
    }
  }
}

__global__ __launch_bounds__(256) void k_out_transpose(
    const float* __restrict__ src, float* __restrict__ dst) {
  __shared__ float tf[32 * 257];
  const int p0  = blockIdx.x * 32;
  const int n   = p0 / HW;
  const int rr0 = p0 % HW;
  {
    int pj = threadIdx.x >> 3;
    int c0 = (threadIdx.x & 7) * 32;
    const float* s = src + (size_t)(p0 + pj) * CH + c0;
    for (int i = 0; i < 32; ++i) tf[pj * 257 + c0 + i] = s[i];
  }
  __syncthreads();
  {
    int pj = threadIdx.x & 31;
    int c0 = threadIdx.x >> 5;
    for (int c = c0; c < CH; c += 8)
      dst[((size_t)n * CH + c) * HW + rr0 + pj] = tf[pj * 257 + c];
  }
}

extern "C" void kernel_launch(void* const* d_in, const int* in_sizes, int n_in,
                              void* d_out, int out_size, void* d_ws, size_t ws_size,
                              hipStream_t stream) {
  const float* x     = (const float*)d_in[0];
  const float* beta1 = (const float*)d_in[1];
  const float* w1    = (const float*)d_in[2];
  const float* bn1g  = (const float*)d_in[3];
  const float* bn1b  = (const float*)d_in[4];
  const float* bn1m  = (const float*)d_in[5];
  const float* bn1v  = (const float*)d_in[6];
  const float* beta2 = (const float*)d_in[7];
  const float* w2    = (const float*)d_in[8];
  const float* bn2g  = (const float*)d_in[9];
  const float* bn2b  = (const float*)d_in[10];
  const float* bn2m  = (const float*)d_in[11];
  const float* bn2v  = (const float*)d_in[12];
  const float* prg   = (const float*)d_in[13];
  const float* prs   = (const float*)d_in[14];
  const float* prz   = (const float*)d_in[15];
  float* out = (float*)d_out;

  char* ws = (char*)d_ws;
  uint8_t* abin1 = (uint8_t*)(ws + OFF_ABIN1);
  uint8_t* abin2 = (uint8_t*)(ws + OFF_ABIN2);
  float*   outr  = (float*)  (ws + OFF_OUTR);
  float*   xnhwc = (float*)  (ws + OFF_XNHWC);
  float*   outn  = (float*)  (ws + OFF_OUTN);
  uint8_t* wf1   = (uint8_t*)(ws + OFF_WF1);
  uint8_t* wf2   = (uint8_t*)(ws + OFF_WF2);
  float*   al1   = (float*)  (ws + OFF_AL1);
  float*   al2   = (float*)  (ws + OFF_AL2);
  uint8_t* zpg   = (uint8_t*)(ws + OFF_ZPG);

  k_zero<<<1, 64, 0, stream>>>(zpg);
  k_binarize<<<NPIX / 32, 256, 0, stream>>>(x, beta1, abin1, xnhwc);
  k_alpha<<<1, 256, 0, stream>>>(w1, al1, CH * 9);
  k_alpha<<<1, 256, 0, stream>>>(w2, al2, CH);
  k_packw<<<(16 * NKT1 * 32 * 8 + 255) / 256, 256, 0, stream>>>(w1, wf1, NKT1, 1);
  k_packw<<<(16 * NKT2 * 32 * 8 + 255) / 256, 256, 0, stream>>>(w2, wf2, NKT2, 0);

  dim3 grid(PGROUPS, 4);
  k_gemm3x3<<<grid, 256, 4 * NKT1 * 1024, stream>>>(
      abin1, wf1, al1, xnhwc, bn1g, bn1b, bn1m, bn1v,
      prg, prs, prz, beta2, zpg, outr, abin2);
  k_gemm1x1<<<grid, 256, 0, stream>>>(
      abin2, wf2, al2, outr, bn2g, bn2b, bn2m, bn2v,
      prg, prs, prz, outn);
  k_out_transpose<<<NPIX / 32, 256, 0, stream>>>(outn, out);
}
